// DSNTDoubleLossNewMSECnoED_20083267076464
// MI455X (gfx1250) — matrix operand order repaired and checked by reading, not compiled, not run on hardware
//
#include <hip/hip_runtime.h>
#include <hip/hip_bf16.h>
#include <stdint.h>

// ---------------------------------------------------------------------------
// DSNT double-loss: B=32, C=8, H=256, W=256  (256 heatmaps of 64K fp32)
// Memory-bound: ~128 MiB mandatory traffic -> ~5.5us at 23.3 TB/s.
// Pass A: per-heatmap max(input) + argmax(target)      (reads both tensors)
// Pass B: WMMA-accumulated softmax moments, async-LDS   (re-reads input, L2 hot)
// Pass C: deterministic final sum -> scalar
// ---------------------------------------------------------------------------

typedef __attribute__((ext_vector_type(2))) float v2f;
typedef __attribute__((ext_vector_type(8))) float v8f;

#if defined(__has_builtin)
# if __has_builtin(__builtin_amdgcn_wmma_f32_16x16x4_f32)
#  define HAVE_WMMA_K4 1
# else
#  define HAVE_WMMA_K4 0
# endif
#else
# define HAVE_WMMA_K4 0
#endif

__device__ __forceinline__ v8f wmma_k4(v2f a, v2f b, v8f c) {
#if HAVE_WMMA_K4
    // D(16x16,f32) = A(16x4,f32) x B(4x16,f32) + C   -- full fp32 precision
    return __builtin_amdgcn_wmma_f32_16x16x4_f32(false, a, false, b,
                                                 (short)0, c, false, false);
#else
    // Fallback: confirmed bf16 WMMA (keeps wmma path live if K4 builtin absent)
    typedef __attribute__((ext_vector_type(16))) __bf16 v16bf;
    v16bf ab, bb;
#pragma unroll
    for (int i = 0; i < 16; ++i) { ab[i] = (__bf16)0.0f; bb[i] = (__bf16)0.0f; }
    ab[0] = (__bf16)a.x; ab[1] = (__bf16)a.y;
    bb[0] = (__bf16)b.x; bb[1] = (__bf16)b.y;
    return __builtin_amdgcn_wmma_f32_16x16x32_bf16(false, ab, false, bb,
                                                   (short)0, c, false, false);
#endif
}

__device__ __forceinline__ void async_b128_to_lds(uint32_t lds_byte_addr,
                                                  const void* gaddr) {
    // VGLOBAL async copy, tracked by ASYNCcnt (CDNA5 ISA)
    asm volatile("global_load_async_to_lds_b128 %0, %1, off"
                 :: "v"(lds_byte_addr), "v"(gaddr) : "memory");
}
__device__ __forceinline__ void wait_async_le4() {
    asm volatile("s_wait_asynccnt 4" ::: "memory");
}
__device__ __forceinline__ void wait_async_le0() {
    asm volatile("s_wait_asynccnt 0" ::: "memory");
}

#define HW 65536            // H*W per heatmap
#define RSTRIDE 260         // dwords per tile row (256 + 4 pad -> bank-conflict free)

__device__ __forceinline__ float coordf(int i) {
    return ((float)i + 1.0f - 128.0f) * (1.0f / 256.0f);
}

// ---------------------------------------------------------------------------
// Pass A: max(input) and argmax(target) per heatmap (coalesced b128 streaming)
// ---------------------------------------------------------------------------
__global__ void __launch_bounds__(256)
dsnt_pass_a(const float* __restrict__ in, const float* __restrict__ tgt,
            float* __restrict__ maxv, int* __restrict__ amax) {
    const int bc = blockIdx.x;
    const int t  = threadIdx.x;
    const float4* ip = (const float4*)(in  + (size_t)bc * HW);
    const float4* tp = (const float4*)(tgt + (size_t)bc * HW);

    float mi = -INFINITY, mt = -INFINITY;
    int mti = 0;
#pragma unroll 4
    for (int j = 0; j < 64; ++j) {
        int q = j * 256 + t;
        float4 a = ip[q];
        float4 b = tp[q];
        mi = fmaxf(mi, fmaxf(fmaxf(a.x, a.y), fmaxf(a.z, a.w)));
        int base = q * 4;
        if (b.x > mt) { mt = b.x; mti = base; }
        if (b.y > mt) { mt = b.y; mti = base + 1; }
        if (b.z > mt) { mt = b.z; mti = base + 2; }
        if (b.w > mt) { mt = b.w; mti = base + 3; }
    }

    __shared__ float smi[256];
    __shared__ float smt[256];
    __shared__ int   sid[256];
    smi[t] = mi; smt[t] = mt; sid[t] = mti;
    __syncthreads();
    for (int off = 128; off > 0; off >>= 1) {
        if (t < off) {
            smi[t] = fmaxf(smi[t], smi[t + off]);
            float vo = smt[t + off]; int io = sid[t + off];
            if (vo > smt[t] || (vo == smt[t] && io < sid[t])) {
                smt[t] = vo; sid[t] = io;   // first-occurrence tie-break
            }
        }
        __syncthreads();
    }
    if (t == 0) { maxv[bc] = smi[0]; amax[bc] = sid[0]; }
}

// ---------------------------------------------------------------------------
// Pass B: softmax moments via WMMA, async double-buffered LDS staging.
// Block = 256 threads (8 waves) per heatmap. 16 stages of 16x256 tiles.
// Wave w owns K-columns [32w, 32w+32) of every stage.
//   C1 += exp(x-m) x [1 | xs | 0...]   -> col0 = S, col1 = S*x
//   C2 += (ys(row)*exp(x-m)) x same    -> col0 = S*y
// ---------------------------------------------------------------------------
__global__ void __launch_bounds__(256)
dsnt_pass_b(const float* __restrict__ in, const float* __restrict__ maxv,
            const int* __restrict__ amax, float* __restrict__ mse) {
    __shared__ float tile[2][16 * RSTRIDE];
    __shared__ float wred[8][3];

    const int bc   = blockIdx.x;
    const int t    = threadIdx.x;
    const int lane = t & 31;
    const int wv   = t >> 5;
    const int n    = lane & 15;     // B/C/D column, also A row
    const int hi   = lane >> 4;     // half-wave selector
    const float* hp = in + (size_t)bc * HW;
    const float  m  = maxv[bc];

    auto issue_stage = [&](int s, int bufsel) {
        uint32_t ldsbase = (uint32_t)(uintptr_t)&tile[bufsel][0];
        const float* gb = hp + s * 16 * 256;
#pragma unroll
        for (int i = 0; i < 4; ++i) {
            int q   = i * 256 + t;      // float4 index within 16x256 tile
            int row = q >> 6;
            int c4  = q & 63;
            uint32_t la = ldsbase + (uint32_t)((row * RSTRIDE + c4 * 4) * 4);
            const float* ga = gb + row * 256 + c4 * 4;
            async_b128_to_lds(la, ga);
        }
    };

    v8f c1 = {0.f, 0.f, 0.f, 0.f, 0.f, 0.f, 0.f, 0.f};
    v8f c2 = {0.f, 0.f, 0.f, 0.f, 0.f, 0.f, 0.f, 0.f};
    const int k0base = wv * 32;

    issue_stage(0, 0);
    for (int s = 0; s < 16; ++s) {
        if (s < 15) { issue_stage(s + 1, (s + 1) & 1); wait_async_le4(); }
        else        { wait_async_le0(); }
        __syncthreads();                         // all waves' stage-s data in LDS

        const float* buf  = &tile[s & 1][0];
        const float* trow = buf + n * RSTRIDE;   // A row = lane&15
        const float  ysr  = coordf(s * 16 + n);  // ys weight for this row
#pragma unroll
        for (int kk = 0; kk < 8; ++kk) {
            int k0 = k0base + kk * 4;
            float2 x2 = *(const float2*)(trow + k0 + 2 * hi);
            float e0 = __expf(x2.x - m);
            float e1 = __expf(x2.y - m);
            v2f a  = {e0, e1};
            v2f a2 = {e0 * ysr, e1 * ysr};
            float xv0 = coordf(k0 + 2 * hi);
            float xv1 = coordf(k0 + 2 * hi + 1);
            float b0 = (n == 0) ? 1.0f : ((n == 1) ? xv0 : 0.0f);
            float b1 = (n == 0) ? 1.0f : ((n == 1) ? xv1 : 0.0f);
            v2f b = {b0, b1};
            c1 = wmma_k4(a,  b, c1);
            c2 = wmma_k4(a2, b, c2);
        }
        __syncthreads();                         // buffer safe to overwrite
    }

    // Extract columns 0 (S, S*y) and 1 (S*x); reduce across wave.
    float s_p = 0.f, sx_p = 0.f, sy_p = 0.f;
    if (n == 0) {
#pragma unroll
        for (int i = 0; i < 8; ++i) { s_p += c1[i]; sy_p += c2[i]; }
    } else if (n == 1) {
#pragma unroll
        for (int i = 0; i < 8; ++i) { sx_p += c1[i]; }
    }
    for (int off = 16; off > 0; off >>= 1) {
        s_p  += __shfl_xor(s_p,  off, 32);
        sx_p += __shfl_xor(sx_p, off, 32);
        sy_p += __shfl_xor(sy_p, off, 32);
    }
    if (lane == 0) { wred[wv][0] = s_p; wred[wv][1] = sx_p; wred[wv][2] = sy_p; }
    __syncthreads();

    if (t == 0) {
        float S = 0.f, SX = 0.f, SY = 0.f;
#pragma unroll
        for (int i = 0; i < 8; ++i) { S += wred[i][0]; SX += wred[i][1]; SY += wred[i][2]; }
        float px = SX / S, py = SY / S;
        int idx = amax[bc];
        float tx = coordf(idx & 255);
        float ty = coordf(idx >> 8);
        float dx = px - tx, dy = py - ty;
        mse[bc] = 0.5f * (dx * dx + dy * dy);
    }
}

// ---------------------------------------------------------------------------
// Pass C: deterministic final sum of 256 per-heatmap losses, / batch(32)
// ---------------------------------------------------------------------------
__global__ void __launch_bounds__(256)
dsnt_pass_c(const float* __restrict__ mse, float* __restrict__ out) {
    __shared__ float s[256];
    const int t = threadIdx.x;
    s[t] = mse[t];
    __syncthreads();
    for (int off = 128; off > 0; off >>= 1) {
        if (t < off) s[t] += s[t + off];
        __syncthreads();
    }
    if (t == 0) out[0] = s[0] * (1.0f / 32.0f);
}

extern "C" void kernel_launch(void* const* d_in, const int* in_sizes, int n_in,
                              void* d_out, int out_size, void* d_ws, size_t ws_size,
                              hipStream_t stream) {
    const float* in  = (const float*)d_in[0];
    const float* tgt = (const float*)d_in[1];
    float* out  = (float*)d_out;
    float* maxv = (float*)d_ws;                            // 256 f32
    int*   amax = (int*)((char*)d_ws + 256 * sizeof(float)); // 256 i32
    float* msep = (float*)((char*)d_ws + 512 * sizeof(float)); // 256 f32

    dsnt_pass_a<<<256, 256, 0, stream>>>(in, tgt, maxv, amax);
    dsnt_pass_b<<<256, 256, 0, stream>>>(in, maxv, amax, msep);
    dsnt_pass_c<<<1, 256, 0, stream>>>(msep, out);
}
